// ST_Block_37589553774692
// MI455X (gfx1250) — matrix operand order repaired and checked by reading, not compiled, not run on hardware
//
#include <hip/hip_runtime.h>
#include <math.h>

// ---------------------------------------------------------------------------
// Types for CDNA5 WMMA (wave32): v16bf A/B fragments, v8f accumulator.
// ---------------------------------------------------------------------------
typedef __bf16 bf16_t;
typedef __attribute__((ext_vector_type(16))) __bf16 v16bf;
typedef __attribute__((ext_vector_type(8)))  __bf16 v8bf;
typedef __attribute__((ext_vector_type(8)))  float  v8f;

#define DEVINL __device__ __forceinline__

DEVINL v8f wmma_bf16(v16bf a, v16bf b, v8f c) {
  // D = A(16x32) * B(32x16) + C(16x16 f32)  -> v_wmma_f32_16x16x32_bf16
  return __builtin_amdgcn_wmma_f32_16x16x32_bf16(false, a, false, b, (short)0, c,
                                                 false, false);
}

DEVINL v16bf pack16(v8bf lo, v8bf hi) {
  v16bf r;
#pragma unroll
  for (int i = 0; i < 8; ++i) { r[i] = lo[i]; r[i + 8] = hi[i]; }
  return r;
}

DEVINL v16bf zero_frag() {
  v16bf r;
#pragma unroll
  for (int i = 0; i < 16; ++i) r[i] = (__bf16)0.0f;
  return r;
}

// A fragment: 16x32 bf16 tile (rows row0..row0+15, K k0..k0+31), A row-major.
// ISA layout: lanes 0-15 hold K {0..7,16..23}, lanes 16-31 hold K {8..15,24..31}.
DEVINL v16bf load_a_frag(const bf16_t* A, int lda, int row0, int k0, int lane) {
  int m  = row0 + (lane & 15);
  int kk = k0 + ((lane & 16) ? 8 : 0);
  const bf16_t* p = A + (long)m * lda + kk;
  v8bf lo = *(const v8bf*)p;
  v8bf hi = *(const v8bf*)(p + 16);
  return pack16(lo, hi);
}

// A fragment with a time shift along rows (row = b*T + t); out-of-range -> zeros.
DEVINL v16bf load_a_frag_shift(const bf16_t* A, int lda, int row0, int k0,
                               int lane, int T, int shift) {
  int m  = row0 + (lane & 15);
  int t  = m % T;
  int ts = t + shift;
  v16bf r = zero_frag();
  if ((unsigned)ts < (unsigned)T) {
    const bf16_t* p = A + (long)(m - t + ts) * lda + k0 + ((lane & 16) ? 8 : 0);
    v8bf lo = *(const v8bf*)p;
    v8bf hi = *(const v8bf*)(p + 16);
    r = pack16(lo, hi);
  }
  return r;
}

// B fragment: 32x16 tile, B supplied TRANSPOSED as Bt[N][K] row-major.
// ISA layout: lanes 0-15 hold K {0..7,16..23}, lanes 16-31 hold K {8..15,24..31} of column n.
DEVINL v16bf load_b_frag(const bf16_t* Bt, int ldb, int col0, int k0, int lane) {
  int n  = col0 + (lane & 15);
  int kk = k0 + ((lane & 16) ? 8 : 0);
  const bf16_t* p = Bt + (long)n * ldb + kk;
  v8bf lo = *(const v8bf*)p;
  v8bf hi = *(const v8bf*)(p + 16);
  return pack16(lo, hi);
}

// C/D layout: VGPR r -> row row0+r (lanes 0-15) / row0+8+r (lanes 16-31), col = lane%16.
DEVINL void store_tile(v8f c, float* C, int ldc, bf16_t* Cb, int ldcb, int transCb,
                       int row0, int col0, const float* bias, float alpha,
                       int relu, int lane) {
  int col = col0 + (lane & 15);
  int rb  = row0 + ((lane & 16) ? 8 : 0);
  float bv = bias ? bias[col] : 0.0f;
#pragma unroll
  for (int r = 0; r < 8; ++r) {
    float v = c[r] * alpha + bv;
    if (relu) v = fmaxf(v, 0.0f);
    int row = rb + r;
    if (C) C[(long)row * ldc + col] = v;
    if (Cb) {
      if (transCb) Cb[(long)col * ldcb + row] = (bf16_t)v;
      else         Cb[(long)row * ldcb + col] = (bf16_t)v;
    }
  }
}

// ---------------------------------------------------------------------------
// Generic batched GEMM:  out[M,N] = act(alpha * A[M,K] @ Bt[N,K]^T + bias)
// Block = 256 thr = 8 waves (2x4). Wave tile 64x32 (4 A-frags x 2 B-frags ->
// 8 WMMA per K-step on 12 b128 loads). Block tile 128x128.
// grid = (N/128, M/128, batch). fp32 and/or bf16 (optionally transposed) output.
// ---------------------------------------------------------------------------
__global__ __launch_bounds__(256) void k_gemm(
    const bf16_t* __restrict__ A, long sA,
    const bf16_t* __restrict__ Bt, long sB,
    float* __restrict__ C, long sC,
    bf16_t* __restrict__ Cb, long sCb, int transCb,
    const float* __restrict__ bias,
    int K, int lda, int ldb, int ldc, int ldcb, float alpha, int relu) {
  int lane = threadIdx.x & 31;
  int wave = threadIdx.x >> 5;
  int bz = blockIdx.z;
  const bf16_t* Ab = A + (long)bz * sA;
  const bf16_t* Bb = Bt + (long)bz * sB;
  float*  Cp  = C  ? C  + (long)bz * sC  : (float*)0;
  bf16_t* Cbp = Cb ? Cb + (long)bz * sCb : (bf16_t*)0;

  int row0 = blockIdx.y * 128 + (wave >> 2) * 64;
  int col0 = blockIdx.x * 128 + (wave & 3) * 32;

  v8f acc[4][2];
#pragma unroll
  for (int i = 0; i < 4; ++i)
#pragma unroll
    for (int j = 0; j < 2; ++j)
#pragma unroll
      for (int e = 0; e < 8; ++e) acc[i][j][e] = 0.f;

  for (int k0 = 0; k0 < K; k0 += 32) {
    v16bf a[4], b[2];
#pragma unroll
    for (int i = 0; i < 4; ++i) a[i] = load_a_frag(Ab, lda, row0 + 16 * i, k0, lane);
#pragma unroll
    for (int j = 0; j < 2; ++j) b[j] = load_b_frag(Bb, ldb, col0 + 16 * j, k0, lane);
#pragma unroll
    for (int i = 0; i < 4; ++i)
#pragma unroll
      for (int j = 0; j < 2; ++j)
        acc[i][j] = wmma_bf16(a[i], b[j], acc[i][j]);
  }
#pragma unroll
  for (int i = 0; i < 4; ++i)
#pragma unroll
    for (int j = 0; j < 2; ++j)
      store_tile(acc[i][j], Cp, ldc, Cbp, ldcb, transCb,
                 row0 + 16 * i, col0 + 16 * j, bias, alpha, relu, lane);
}

// ---------------------------------------------------------------------------
// Dilated causal/anticausal conv (kernel=2) as a dual-tap GEMM:
//   out[bt, co] = relu(bias[co] + A[bt+shift] @ W0t^T + A[bt] @ W1t^T)
// A rows indexed by bt = b*T + t; shifted rows out of [0,T) contribute zeros.
// Wave tile 64x32, 16 WMMA per K-step. grid = (Nc/128, (B*T)/128, 1)
// ---------------------------------------------------------------------------
__global__ __launch_bounds__(256) void k_conv_gemm(
    const bf16_t* __restrict__ A, int lda,
    const bf16_t* __restrict__ W0t, const bf16_t* __restrict__ W1t,
    const float* __restrict__ bias,
    float* __restrict__ C, bf16_t* __restrict__ Cb,
    int Nc, int K, int T, int shift) {
  int lane = threadIdx.x & 31;
  int wave = threadIdx.x >> 5;
  int row0 = blockIdx.y * 128 + (wave >> 2) * 64;
  int col0 = blockIdx.x * 128 + (wave & 3) * 32;

  v8f acc[4][2];
#pragma unroll
  for (int i = 0; i < 4; ++i)
#pragma unroll
    for (int j = 0; j < 2; ++j)
#pragma unroll
      for (int e = 0; e < 8; ++e) acc[i][j][e] = 0.f;

  for (int k0 = 0; k0 < K; k0 += 32) {
    v16bf as[4], a[4], w0[2], w1[2];
#pragma unroll
    for (int i = 0; i < 4; ++i) {
      as[i] = load_a_frag_shift(A, lda, row0 + 16 * i, k0, lane, T, shift);
      a[i]  = load_a_frag(A, lda, row0 + 16 * i, k0, lane);
    }
#pragma unroll
    for (int j = 0; j < 2; ++j) {
      w0[j] = load_b_frag(W0t, K, col0 + 16 * j, k0, lane);
      w1[j] = load_b_frag(W1t, K, col0 + 16 * j, k0, lane);
    }
#pragma unroll
    for (int i = 0; i < 4; ++i)
#pragma unroll
      for (int j = 0; j < 2; ++j) {
        acc[i][j] = wmma_bf16(as[i], w0[j], acc[i][j]);
        acc[i][j] = wmma_bf16(a[i],  w1[j], acc[i][j]);
      }
  }
#pragma unroll
  for (int i = 0; i < 4; ++i)
#pragma unroll
    for (int j = 0; j < 2; ++j)
      store_tile(acc[i][j], C, Nc, Cb, Nc, 0,
                 row0 + 16 * i, col0 + 16 * j, bias, 1.0f, 1, lane);
}

// ---------------------------------------------------------------------------
// Row-wise kernels (wave32 per 128-wide row, 4 values per lane)
// ---------------------------------------------------------------------------
__global__ __launch_bounds__(256) void k_topk_softmax(
    const float* __restrict__ att, bf16_t* __restrict__ spb, int rows) {
  int lane = threadIdx.x & 31;
  long wrow = (long)blockIdx.x * (blockDim.x >> 5) + (threadIdx.x >> 5);
  if (wrow >= rows) return;
  const float* rp = att + wrow * 128;
  float v[4], w0[4];
#pragma unroll
  for (int j = 0; j < 4; ++j) { v[j] = rp[lane * 4 + j]; w0[j] = v[j]; }
  float thr = 0.f, rowmax = 0.f;
  for (int it = 0; it < 16; ++it) {
    float m = fmaxf(fmaxf(w0[0], w0[1]), fmaxf(w0[2], w0[3]));
    for (int s = 16; s > 0; s >>= 1) m = fmaxf(m, __shfl_xor(m, s, 32));
    if (it == 0) rowmax = m;
    thr = m;
    bool has = (w0[0] == m) || (w0[1] == m) || (w0[2] == m) || (w0[3] == m);
    unsigned long long bal = __ballot(has);
    int leader = __ffsll(bal) - 1;
    if (lane == leader) {
      if      (w0[0] == m) w0[0] = -INFINITY;
      else if (w0[1] == m) w0[1] = -INFINITY;
      else if (w0[2] == m) w0[2] = -INFINITY;
      else                 w0[3] = -INFINITY;
    }
  }
  float e[4], sum = 0.f;
#pragma unroll
  for (int j = 0; j < 4; ++j) { e[j] = (v[j] >= thr) ? __expf(v[j] - rowmax) : 0.f; sum += e[j]; }
  for (int s = 16; s > 0; s >>= 1) sum += __shfl_xor(sum, s, 32);
  float inv = (sum > 0.f) ? 1.0f / sum : 0.0f;
#pragma unroll
  for (int j = 0; j < 4; ++j) spb[wrow * 128 + lane * 4 + j] = (bf16_t)(e[j] * inv);
}

__global__ __launch_bounds__(256) void k_mask_softmax(
    const float* __restrict__ gm, bf16_t* __restrict__ outb, int rows) {
  int lane = threadIdx.x & 31;
  long wrow = (long)blockIdx.x * (blockDim.x >> 5) + (threadIdx.x >> 5);
  if (wrow >= rows) return;
  const float* rp = gm + wrow * 128;
  float v[4]; bool use[4];
  float m = -INFINITY;
#pragma unroll
  for (int j = 0; j < 4; ++j) {
    v[j] = rp[lane * 4 + j];
    use[j] = (v[j] != 0.0f);
    if (use[j]) m = fmaxf(m, v[j]);
  }
  for (int s = 16; s > 0; s >>= 1) m = fmaxf(m, __shfl_xor(m, s, 32));
  float e[4], sum = 0.f;
#pragma unroll
  for (int j = 0; j < 4; ++j) { e[j] = use[j] ? __expf(v[j] - m) : 0.0f; sum += e[j]; }
  for (int s = 16; s > 0; s >>= 1) sum += __shfl_xor(sum, s, 32);
  float inv = (sum > 0.f) ? 1.0f / sum : 0.0f;
#pragma unroll
  for (int j = 0; j < 4; ++j) outb[wrow * 128 + lane * 4 + j] = (bf16_t)(e[j] * inv);
}

// ---------------------------------------------------------------------------
// Elementwise / repack kernels
// ---------------------------------------------------------------------------
__global__ void k_convert(const float* __restrict__ s, bf16_t* __restrict__ d, long n) {
  long i = (long)blockIdx.x * blockDim.x + threadIdx.x;
  if (i < n) d[i] = (bf16_t)s[i];
}

// fp32 src[Mr][Nc] -> bf16 dst[Nc][Mr] (weight transpose: [in,out] -> Bt=[out,in])
__global__ void k_transpose_w(const float* __restrict__ src, bf16_t* __restrict__ dst,
                              int Mr, int Nc) {
  long i = (long)blockIdx.x * blockDim.x + threadIdx.x;
  long total = (long)Mr * Nc;
  if (i >= total) return;
  int n = (int)(i % Nc);
  int m = (int)(i / Nc);
  dst[(long)n * Mr + m] = (bf16_t)src[i];
}

// conv weights fp32 [L][Co][Ci][2] -> bf16 [L*2 + k][Co][Ci]  (de-interleave taps)
__global__ void k_convw(const float* __restrict__ W, bf16_t* __restrict__ dst,
                        int L, int Co, int Ci) {
  long i = (long)blockIdx.x * blockDim.x + threadIdx.x;
  long total = (long)L * Co * Ci * 2;
  if (i >= total) return;
  int k = (int)(i & 1);
  long j = i >> 1;
  int ci = (int)(j % Ci); j /= Ci;
  int co = (int)(j % Co);
  int l  = (int)(j / Co);
  dst[(((long)l * 2 + k) * Co + co) * Ci + ci] = (bf16_t)W[i];
}

// x[b][t][n][d] -> xf[b][n][t*16+d] (fp32 + bf16)
__global__ void k_build_xf(const float* __restrict__ x, float* __restrict__ xf,
                           bf16_t* __restrict__ xfb) {
  long i = (long)blockIdx.x * blockDim.x + threadIdx.x;
  const long total = 32L * 48 * 128 * 16;
  if (i >= total) return;
  int d = (int)(i & 15);
  long j = i >> 4;
  int n = (int)(j & 127); j >>= 7;
  int t = (int)(j % 48);
  int b = (int)(j / 48);
  long o = ((long)b * 128 + n) * 768 + t * 16 + d;
  float v = x[i];
  xf[o] = v;
  xfb[o] = (bf16_t)v;
}

// xs[b][t][n*16+d] = x + sparse_out[b][n][t*16+d] + g2[b][n][t*16+d]  (fp32 + bf16)
__global__ void k_make_xs(const float* __restrict__ x, const float* __restrict__ so,
                          const float* __restrict__ g2, float* __restrict__ xs,
                          bf16_t* __restrict__ xsb) {
  long i = (long)blockIdx.x * blockDim.x + threadIdx.x;
  const long total = 32L * 48 * 128 * 16;
  if (i >= total) return;
  int d = (int)(i & 15);
  long j = i >> 4;
  int n = (int)(j & 127); j >>= 7;
  int t = (int)(j % 48);
  int b = (int)(j / 48);
  long o = ((long)b * 128 + n) * 768 + t * 16 + d;
  float v = x[i] + so[o] + g2[o];
  xs[i] = v;
  xsb[i] = (bf16_t)v;
}

__global__ void k_add3_bf16(const float* __restrict__ a, const float* __restrict__ b,
                            const float* __restrict__ c, bf16_t* __restrict__ d, long n) {
  long i = (long)blockIdx.x * blockDim.x + threadIdx.x;
  if (i < n) d[i] = (bf16_t)(a[i] + b[i] + c[i]);
}

// out[b][t][n][d] = (n<80) ? fF+rF at [bt][n*16+d] : fS+rS at [bt][(n-80)*16+d]
__global__ void k_final(const float* __restrict__ fF, const float* __restrict__ rF,
                        const float* __restrict__ fS, const float* __restrict__ rS,
                        float* __restrict__ out) {
  long i = (long)blockIdx.x * blockDim.x + threadIdx.x;
  const long total = 32L * 48 * 128 * 16;
  if (i >= total) return;
  int d = (int)(i & 15);
  long j = i >> 4;
  int n = (int)(j & 127); j >>= 7;
  int t = (int)(j % 48);
  int b = (int)(j / 48);
  long bt = (long)b * 48 + t;
  float v;
  if (n < 80) {
    long o = bt * 1280 + n * 16 + d;
    v = fF[o] + rF[o];
  } else {
    long o = bt * 768 + (n - 80) * 16 + d;
    v = fS[o] + rS[o];
  }
  out[i] = v;
}

// ---------------------------------------------------------------------------
// Host orchestration
// ---------------------------------------------------------------------------
extern "C" void kernel_launch(void* const* d_in, const int* in_sizes, int n_in,
                              void* d_out, int out_size, void* d_ws, size_t ws_size,
                              hipStream_t stream) {
  (void)in_sizes; (void)n_in; (void)out_size; (void)ws_size;

  const float* x      = (const float*)d_in[0];
  const float* graph  = (const float*)d_in[1];
  const float* qw     = (const float*)d_in[2];
  const float* qb     = (const float*)d_in[3];
  const float* kw     = (const float*)d_in[4];
  const float* kb     = (const float*)d_in[5];
  const float* vw     = (const float*)d_in[6];
  const float* vb     = (const float*)d_in[7];
  const float* ow     = (const float*)d_in[8];
  const float* ob     = (const float*)d_in[9];
  const float* l1w    = (const float*)d_in[10];
  const float* l1b    = (const float*)d_in[11];
  const float* l2w    = (const float*)d_in[12];
  const float* l2b    = (const float*)d_in[13];
  const float* btcnW  = (const float*)d_in[14];
  const float* btcnB  = (const float*)d_in[15];
  const float* btcnfW = (const float*)d_in[16];
  const float* btcnfB = (const float*)d_in[17];
  const float* btcnsW = (const float*)d_in[18];
  const float* btcnsB = (const float*)d_in[19];
  float* out = (float*)d_out;

  const int Bn = 32, Tn = 48, Nn = 128, Dm = 768;
  const int Cc = 2048, CF = 1280, CS = 768;
  const int BT = Bn * Tn;      // 1536
  const int BN = Bn * Nn;      // 4096
  const float att_scale = 1.0f / sqrtf((float)Dm);

  // bump allocator over workspace
  size_t off = 0;
  char* base = (char*)d_ws;
  auto alloc = [&](size_t bytes) -> char* {
    off = (off + 255) & ~(size_t)255;
    char* p = base + off;
    off += bytes;
    return p;
  };

  // bf16 weight repacks (transposed Bt = [out,in]); fits in 192MB L2 after first touch
  bf16_t* qwT  = (bf16_t*)alloc((size_t)768 * 768 * 2);
  bf16_t* kwT  = (bf16_t*)alloc((size_t)768 * 768 * 2);
  bf16_t* vwT  = (bf16_t*)alloc((size_t)768 * 768 * 2);
  bf16_t* owT  = (bf16_t*)alloc((size_t)768 * 768 * 2);
  bf16_t* l1wT = (bf16_t*)alloc((size_t)3072 * 768 * 2);
  bf16_t* l2wT = (bf16_t*)alloc((size_t)768 * 3072 * 2);
  bf16_t* grB  = (bf16_t*)alloc((size_t)128 * 128 * 2);
  bf16_t* Wm   = (bf16_t*)alloc((size_t)4 * 2 * Cc * Cc * 2);
  bf16_t* Wf   = (bf16_t*)alloc((size_t)4 * 2 * CF * CF * 2);
  bf16_t* Ws   = (bf16_t*)alloc((size_t)4 * 2 * CS * CS * 2);

  // activations
  float*  xf   = (float*) alloc((size_t)BN * Dm * 4);
  bf16_t* xfB  = (bf16_t*)alloc((size_t)BN * Dm * 2);
  bf16_t* qB   = (bf16_t*)alloc((size_t)BN * Dm * 2);
  bf16_t* kB   = (bf16_t*)alloc((size_t)BN * Dm * 2);
  bf16_t* vTB  = (bf16_t*)alloc((size_t)Bn * Dm * Nn * 2);  // [b][768][128]
  float*  att  = (float*) alloc((size_t)Bn * Nn * Nn * 4);
  bf16_t* attB = (bf16_t*)alloc((size_t)Bn * Nn * Nn * 2);
  bf16_t* spB  = (bf16_t*)alloc((size_t)Bn * Nn * Nn * 2);
  bf16_t* s1B  = (bf16_t*)alloc((size_t)BN * Dm * 2);
  float*  so   = (float*) alloc((size_t)BN * Dm * 4);
  float*  gm   = (float*) alloc((size_t)Bn * Nn * Nn * 4);
  bf16_t* gmB  = (bf16_t*)alloc((size_t)Bn * Nn * Nn * 2);
  bf16_t* h1TB = (bf16_t*)alloc((size_t)Bn * 3072 * Nn * 2); // [b][3072][128]
  bf16_t* g1B  = (bf16_t*)alloc((size_t)BN * 3072 * 2);
  bf16_t* h2TB = (bf16_t*)alloc((size_t)Bn * Dm * Nn * 2);   // [b][768][128]
  float*  g2   = (float*) alloc((size_t)BN * Dm * 4);
  float*  xs   = (float*) alloc((size_t)BT * Cc * 4);
  bf16_t* xsB  = (bf16_t*)alloc((size_t)BT * Cc * 2);
  bf16_t* tB   = (bf16_t*)alloc((size_t)BT * Cc * 2);
  float*  pf[6]; bf16_t* pb[6];
  for (int i = 0; i < 6; ++i) {
    pf[i] = (float*) alloc((size_t)BT * Cc * 4);
    pb[i] = (bf16_t*)alloc((size_t)BT * Cc * 2);
  }

  auto ceil_div = [](long a, long b) { return (int)((a + b - 1) / b); };

  // ---- weight repack ----
  k_transpose_w<<<ceil_div(768L * 768, 256), 256, 0, stream>>>(qw, qwT, 768, 768);
  k_transpose_w<<<ceil_div(768L * 768, 256), 256, 0, stream>>>(kw, kwT, 768, 768);
  k_transpose_w<<<ceil_div(768L * 768, 256), 256, 0, stream>>>(vw, vwT, 768, 768);
  k_transpose_w<<<ceil_div(768L * 768, 256), 256, 0, stream>>>(ow, owT, 768, 768);
  k_transpose_w<<<ceil_div(768L * 3072, 256), 256, 0, stream>>>(l1w, l1wT, 768, 3072);
  k_transpose_w<<<ceil_div(3072L * 768, 256), 256, 0, stream>>>(l2w, l2wT, 3072, 768);
  k_convert<<<ceil_div(128L * 128, 256), 256, 0, stream>>>(graph, grB, 128L * 128);
  k_convw<<<ceil_div(4L * Cc * Cc * 2, 256), 256, 0, stream>>>(btcnW,  Wm, 4, Cc, Cc);
  k_convw<<<ceil_div(4L * CF * CF * 2, 256), 256, 0, stream>>>(btcnfW, Wf, 4, CF, CF);
  k_convw<<<ceil_div(4L * CS * CS * 2, 256), 256, 0, stream>>>(btcnsW, Ws, 4, CS, CS);

  // ---- spatial ----
  k_build_xf<<<ceil_div((long)BN * Dm, 256), 256, 0, stream>>>(x, xf, xfB);

  // q,k (bf16 only), v (transposed bf16, batched)
  k_gemm<<<dim3(6, 32, 1), 256, 0, stream>>>(xfB, 0, qwT, 0, (float*)0, 0, qB,
      0, 0, qb, 768, 768, 768, 0, 768, 1.0f, 0);
  k_gemm<<<dim3(6, 32, 1), 256, 0, stream>>>(xfB, 0, kwT, 0, (float*)0, 0, kB,
      0, 0, kb, 768, 768, 768, 0, 768, 1.0f, 0);
  k_gemm<<<dim3(6, 1, 32), 256, 0, stream>>>(xfB, (long)Nn * Dm, vwT, 0, (float*)0, 0,
      vTB, (long)Dm * Nn, 1, vb, 768, 768, 768, 0, 128, 1.0f, 0);

  // att = scale * q @ k^T  (fp32 + bf16)
  k_gemm<<<dim3(1, 1, 32), 256, 0, stream>>>(qB, (long)Nn * Dm, kB, (long)Nn * Dm,
      att, (long)Nn * Nn, attB, (long)Nn * Nn, 0, (const float*)0,
      768, 768, 768, 128, 128, att_scale, 0);

  // top-16 threshold + masked softmax -> sp (bf16)
  k_topk_softmax<<<512, 256, 0, stream>>>(att, spB, BN);

  // s1 = sp @ v  (bf16)
  k_gemm<<<dim3(6, 1, 32), 256, 0, stream>>>(spB, (long)Nn * Nn, vTB, (long)Dm * Nn,
      (float*)0, 0, s1B, (long)Nn * Dm, 0, (const float*)0,
      128, 128, 128, 0, 768, 1.0f, 0);

  // sparse_out = s1 @ ow + ob  (fp32)
  k_gemm<<<dim3(6, 32, 1), 256, 0, stream>>>(s1B, 0, owT, 0, so, 0, (bf16_t*)0,
      0, 0, ob, 768, 768, 768, 768, 0, 1.0f, 0);

  // gm = graph @ att^T  (fp32)
  k_gemm<<<dim3(1, 1, 32), 256, 0, stream>>>(grB, 0, attB, (long)Nn * Nn,
      gm, (long)Nn * Nn, (bf16_t*)0, 0, 0, (const float*)0,
      128, 128, 128, 128, 0, 1.0f, 0);
  k_mask_softmax<<<512, 256, 0, stream>>>(gm, gmB, BN);

  // h1 = xf @ l1w + l1b  -> transposed bf16 per batch [b][3072][128]
  k_gemm<<<dim3(24, 1, 32), 256, 0, stream>>>(xfB, (long)Nn * Dm, l1wT, 0,
      (float*)0, 0, h1TB, (long)3072 * Nn, 1, l1b, 768, 768, 768, 0, 128, 1.0f, 0);
  // g1 = relu(gm @ h1)  (bf16)
  k_gemm<<<dim3(24, 1, 32), 256, 0, stream>>>(gmB, (long)Nn * Nn, h1TB, (long)3072 * Nn,
      (float*)0, 0, g1B, (long)Nn * 3072, 0, (const float*)0,
      128, 128, 128, 0, 3072, 1.0f, 1);
  // h2 = g1 @ l2w + l2b  -> transposed bf16 per batch [b][768][128]
  k_gemm<<<dim3(6, 1, 32), 256, 0, stream>>>(g1B, (long)Nn * 3072, l2wT, 0,
      (float*)0, 0, h2TB, (long)Dm * Nn, 1, l2b, 3072, 3072, 3072, 0, 128, 1.0f, 0);
  // g2 = relu(gm @ h2)  (fp32)
  k_gemm<<<dim3(6, 1, 32), 256, 0, stream>>>(gmB, (long)Nn * Nn, h2TB, (long)Dm * Nn,
      g2, (long)Nn * Dm, (bf16_t*)0, 0, 0, (const float*)0,
      128, 128, 128, 768, 0, 1.0f, 1);

  // xs = x + spatial_out   (fp32 + bf16, [B,T,2048] layout)
  k_make_xs<<<ceil_div((long)BT * Cc, 256), 256, 0, stream>>>(x, so, g2, xs, xsB);

  // ---- BTCN main (C=2048), dils {1,2,4,4}; fwd shift=-d, rev shift=+d ----
  auto WmL = [&](int l, int k) { return Wm + ((size_t)(l * 2 + k)) * Cc * Cc; };
  auto WfL = [&](int l, int k) { return Wf + ((size_t)(l * 2 + k)) * CF * CF; };
  auto WsL = [&](int l, int k) { return Ws + ((size_t)(l * 2 + k)) * CS * CS; };
  dim3 gridC(Cc / 128, BT / 128, 1), gridF(CF / 128, BT / 128, 1), gridS(CS / 128, BT / 128, 1);

  // forward chain -> pf[3]
  k_conv_gemm<<<gridC, 256, 0, stream>>>(xsB,   Cc, WmL(0,0), WmL(0,1), btcnB + 0*Cc, (float*)0, pb[0], Cc, Cc, Tn, -1);
  k_conv_gemm<<<gridC, 256, 0, stream>>>(pb[0], Cc, WmL(1,0), WmL(1,1), btcnB + 1*Cc, (float*)0, pb[1], Cc, Cc, Tn, -2);
  k_conv_gemm<<<gridC, 256, 0, stream>>>(pb[1], Cc, WmL(2,0), WmL(2,1), btcnB + 2*Cc, (float*)0, pb[2], Cc, Cc, Tn, -4);
  k_conv_gemm<<<gridC, 256, 0, stream>>>(pb[2], Cc, WmL(3,0), WmL(3,1), btcnB + 3*Cc, pf[3], (bf16_t*)0, Cc, Cc, Tn, -4);
  // reverse chain -> pf[4]
  k_conv_gemm<<<gridC, 256, 0, stream>>>(xsB,   Cc, WmL(0,0), WmL(0,1), btcnB + 0*Cc, (float*)0, pb[0], Cc, Cc, Tn, +1);
  k_conv_gemm<<<gridC, 256, 0, stream>>>(pb[0], Cc, WmL(1,0), WmL(1,1), btcnB + 1*Cc, (float*)0, pb[1], Cc, Cc, Tn, +2);
  k_conv_gemm<<<gridC, 256, 0, stream>>>(pb[1], Cc, WmL(2,0), WmL(2,1), btcnB + 2*Cc, (float*)0, pb[2], Cc, Cc, Tn, +4);
  k_conv_gemm<<<gridC, 256, 0, stream>>>(pb[2], Cc, WmL(3,0), WmL(3,1), btcnB + 3*Cc, pf[4], (bf16_t*)0, Cc, Cc, Tn, +4);

  // t = fwd + rev + xs  (bf16 only; feeds both branches)
  k_add3_bf16<<<ceil_div((long)BT * Cc, 256), 256, 0, stream>>>(pf[3], pf[4], xs, tB, (long)BT * Cc);

  // ---- F branch (CF=1280): A = tB[:, 0:1280], lda=2048 ----
  k_conv_gemm<<<gridF, 256, 0, stream>>>(tB,    Cc, WfL(0,0), WfL(0,1), btcnfB + 0*CF, (float*)0, pb[0], CF, CF, Tn, -1);
  k_conv_gemm<<<gridF, 256, 0, stream>>>(pb[0], CF, WfL(1,0), WfL(1,1), btcnfB + 1*CF, (float*)0, pb[1], CF, CF, Tn, -2);
  k_conv_gemm<<<gridF, 256, 0, stream>>>(pb[1], CF, WfL(2,0), WfL(2,1), btcnfB + 2*CF, (float*)0, pb[0], CF, CF, Tn, -4);
  k_conv_gemm<<<gridF, 256, 0, stream>>>(pb[0], CF, WfL(3,0), WfL(3,1), btcnfB + 3*CF, pf[1], (bf16_t*)0, CF, CF, Tn, -4);
  k_conv_gemm<<<gridF, 256, 0, stream>>>(tB,    Cc, WfL(0,0), WfL(0,1), btcnfB + 0*CF, (float*)0, pb[2], CF, CF, Tn, +1);
  k_conv_gemm<<<gridF, 256, 0, stream>>>(pb[2], CF, WfL(1,0), WfL(1,1), btcnfB + 1*CF, (float*)0, pb[3], CF, CF, Tn, +2);
  k_conv_gemm<<<gridF, 256, 0, stream>>>(pb[3], CF, WfL(2,0), WfL(2,1), btcnfB + 2*CF, (float*)0, pb[2], CF, CF, Tn, +4);
  k_conv_gemm<<<gridF, 256, 0, stream>>>(pb[2], CF, WfL(3,0), WfL(3,1), btcnfB + 3*CF, pf[3], (bf16_t*)0, CF, CF, Tn, +4);

  // ---- S branch (CS=768): A = tB[:, 1280:2048], lda=2048 ----
  const bf16_t* tS = tB + CF;
  k_conv_gemm<<<gridS, 256, 0, stream>>>(tS,    Cc, WsL(0,0), WsL(0,1), btcnsB + 0*CS, (float*)0, pb[0], CS, CS, Tn, -1);
  k_conv_gemm<<<gridS, 256, 0, stream>>>(pb[0], CS, WsL(1,0), WsL(1,1), btcnsB + 1*CS, (float*)0, pb[4], CS, CS, Tn, -2);
  k_conv_gemm<<<gridS, 256, 0, stream>>>(pb[4], CS, WsL(2,0), WsL(2,1), btcnsB + 2*CS, (float*)0, pb[0], CS, CS, Tn, -4);
  k_conv_gemm<<<gridS, 256, 0, stream>>>(pb[0], CS, WsL(3,0), WsL(3,1), btcnsB + 3*CS, pf[4], (bf16_t*)0, CS, CS, Tn, -4);
  k_conv_gemm<<<gridS, 256, 0, stream>>>(tS,    Cc, WsL(0,0), WsL(0,1), btcnsB + 0*CS, (float*)0, pb[0], CS, CS, Tn, +1);
  k_conv_gemm<<<gridS, 256, 0, stream>>>(pb[0], CS, WsL(1,0), WsL(1,1), btcnsB + 1*CS, (float*)0, pb[5], CS, CS, Tn, +2);
  k_conv_gemm<<<gridS, 256, 0, stream>>>(pb[5], CS, WsL(2,0), WsL(2,1), btcnsB + 2*CS, (float*)0, pb[0], CS, CS, Tn, +4);
  k_conv_gemm<<<gridS, 256, 0, stream>>>(pb[0], CS, WsL(3,0), WsL(3,1), btcnsB + 3*CS, pf[5], (bf16_t*)0, CS, CS, Tn, +4);

  // ---- final concat: out[b,t,n,d] ----
  k_final<<<ceil_div((long)Bn * Tn * Nn * 16, 256), 256, 0, stream>>>(
      pf[1], pf[3], pf[4], pf[5], out);
}
